// GATConv_GNN_47828755808396
// MI455X (gfx1250) — compile-verified, matrix operand order checked by reading, not run on hardware
//
#include <hip/hip_runtime.h>
#include <math.h>

// ---------------------------------------------------------------------------
// GATConv GNN forward for MI455X (gfx1250), wave32.
// Dense projections: v_wmma_f32_16x16x32_bf16, A via b128 global loads,
// B pre-packed to WMMA lane layout and staged in LDS (ds_load_b128).
// Edge softmax/aggregation: L2-resident gather/scatter with f32 atomics.
// ---------------------------------------------------------------------------

typedef __attribute__((ext_vector_type(16))) __bf16 bf16x16;
typedef __attribute__((ext_vector_type(8)))  __bf16 bf16x8;
typedef __attribute__((ext_vector_type(8)))  float  f32x8;

#define Nn 50000
#define Ee 800000
#define FIN 128          // K is 128 for all three conv layers
#define FE  16
#define Gg  128
#define Cc  32

__device__ __forceinline__ float lrelu(float x, float s) { return x >= 0.f ? x : s * x; }

// float atomic max via ordered-int trick (buffer initialized to -inf)
__device__ __forceinline__ void atomicMaxF(float* a, float v) {
    if (v >= 0.f) atomicMax((int*)a, __float_as_int(v));
    else          atomicMin((unsigned int*)a, __float_as_uint(v));
}

__device__ __forceinline__ void atomAddF(float* a, float v) {
    unsafeAtomicAdd(a, v);   // lowers to global_atomic_add_f32
}

__device__ __forceinline__ bf16x16 cat16(bf16x8 lo, bf16x8 hi) {
    return __builtin_shufflevector(lo, hi, 0, 1, 2, 3, 4, 5, 6, 7,
                                           8, 9, 10, 11, 12, 13, 14, 15);
}

// ------------------------------- utility -----------------------------------
__global__ void k_fill_f32(float* p, float v, int n) {
    int i = blockIdx.x * blockDim.x + threadIdx.x;
    if (i < n) p[i] = v;
}
__global__ void k_fill_i32(int* p, int v, int n) {
    int i = blockIdx.x * blockDim.x + threadIdx.x;
    if (i < n) p[i] = v;
}
__global__ void k_cvt_bf16(const float* __restrict__ in, __bf16* __restrict__ out, int n) {
    int i = blockIdx.x * blockDim.x + threadIdx.x;
    if (i < n) out[i] = (__bf16)in[i];
}

// --------------------- self-loop edge attr (fill='mean') --------------------
__global__ void k_deg(const int* __restrict__ dst, int* __restrict__ deg) {
    int e = blockIdx.x * blockDim.x + threadIdx.x;
    if (e < Ee) atomicAdd(&deg[dst[e]], 1);
}
__global__ void k_loop_acc(const int* __restrict__ dst, const float* __restrict__ ea,
                           float* __restrict__ lattr) {
    int i = blockIdx.x * blockDim.x + threadIdx.x;
    if (i >= Ee * FE) return;
    int e = i / FE, f = i % FE;
    atomAddF(&lattr[dst[e] * FE + f], ea[i]);
}
__global__ void k_loop_div(float* __restrict__ lattr, const int* __restrict__ deg) {
    int i = blockIdx.x * blockDim.x + threadIdx.x;
    if (i >= Nn * FE) return;
    lattr[i] = lattr[i] / fmaxf((float)deg[i / FE], 1.f);
}

// ----------------------- B packing to WMMA lane layout ----------------------
// Source: W f32 row-major [K=128, Nc].  Dest (bf16):
//   flat = ((tn*KT + kt)*32 + lane)*16 + e
//   col  = tn*16 + (lane&15);  kb = (lane&16)?8:0
//   k    = kt*32 + kb + e + (e>=8 ? 8 : 0)
__global__ void k_pack_B(const float* __restrict__ W, __bf16* __restrict__ Bp, int Nc) {
    int o = blockIdx.x * blockDim.x + threadIdx.x;
    int total = FIN * Nc;
    if (o >= total) return;
    const int KT = FIN / 32;
    int e    = o & 15;
    int lane = (o >> 4) & 31;
    int kt   = (o >> 9) % KT;
    int tn   = o / (KT * 512);
    int col  = tn * 16 + (lane & 15);
    int kb   = (lane & 16) ? 8 : 0;
    int k    = kt * 32 + kb + e + ((e >= 8) ? 8 : 0);
    Bp[o] = (__bf16)W[k * Nc + col];
}

// -------------------------- WMMA bf16 GEMM ----------------------------------
// D[M, NT*16] = A[M,128] (bf16 row-major) * packed-B.  One wave per 16-row
// M-tile computes all NT column tiles (A fragment reused NT times).  Packed B
// staged once per block in LDS.
template <int NT>
__global__ void k_gemm_bf16(const __bf16* __restrict__ A, const __bf16* __restrict__ Bp,
                            float* __restrict__ D, int M) {
    constexpr int Nc = NT * 16;
    constexpr int KT = FIN / 32;
    __shared__ __bf16 sB[NT * KT * 32 * 16];   // == FIN*Nc bf16

    // cooperative copy of packed B into LDS (16B per thread per iter)
    {
        const uint4* src = (const uint4*)Bp;
        uint4* dstp = (uint4*)sB;
        const int nvec = (NT * KT * 32 * 16) / 8;   // 8 bf16 per uint4
        for (int i = threadIdx.x; i < nvec; i += blockDim.x) dstp[i] = src[i];
    }
    __syncthreads();

    int wave = threadIdx.x >> 5;
    int lane = threadIdx.x & 31;
    int tm = blockIdx.x * (blockDim.x >> 5) + wave;
    int nTm = (M + 15) >> 4;
    if (tm >= nTm) return;                     // wave-uniform

    int row0 = tm << 4;
    int rc = lane & 15;
    int kb = (lane & 16) ? 8 : 0;
    int ar = row0 + rc; if (ar >= M) ar = M - 1;

    f32x8 acc[NT];
#pragma unroll
    for (int t = 0; t < NT; ++t) acc[t] = (f32x8){};

#pragma unroll
    for (int kt = 0; kt < KT; ++kt) {
        const __bf16* ap = A + (long)ar * FIN + kt * 32 + kb;
        bf16x16 av = cat16(*(const bf16x8*)ap, *(const bf16x8*)(ap + 16));
#pragma unroll
        for (int tn = 0; tn < NT; ++tn) {
            const __bf16* bl = sB + ((tn * KT + kt) * 32 + lane) * 16;
            bf16x16 bv = cat16(*(const bf16x8*)bl, *(const bf16x8*)(bl + 8));
            acc[tn] = __builtin_amdgcn_wmma_f32_16x16x32_bf16(
                false, av, false, bv, (short)0, acc[tn], false, false);
        }
    }

    int rbase = row0 + ((lane & 16) ? 8 : 0);
#pragma unroll
    for (int tn = 0; tn < NT; ++tn) {
        int cc = tn * 16 + rc;
#pragma unroll
        for (int j = 0; j < 8; ++j) {
            int r = rbase + j;
            if (r < M) D[(long)r * Nc + cc] = acc[tn][j];
        }
    }
}

// ---------------- attention logit pieces (collapsed algebra) ----------------
// V[f,h] = sum_c W_edge[f, h*C+c] * att_edge[h,c]   (FE x H, tiny)
__global__ void k_edge_vec(const float* __restrict__ We, const float* __restrict__ ae,
                           float* __restrict__ V, int H) {
    int i = threadIdx.x;
    if (i >= FE * H) return;
    int f = i / H, h = i % H;
    float s = 0.f;
    for (int c = 0; c < Cc; ++c) s += We[f * (H * Cc) + h * Cc + c] * ae[h * Cc + c];
    V[f * H + h] = s;
}
// a_edge[e,h] = ea[e,:] . V[:,h]
__global__ void k_a_edge(const float* __restrict__ ea, const float* __restrict__ V,
                         float* __restrict__ out, int rows, int H) {
    int i = blockIdx.x * blockDim.x + threadIdx.x;
    if (i >= rows * H) return;
    int e = i / H, h = i % H;
    float s = 0.f;
    for (int f = 0; f < FE; ++f) s += ea[e * FE + f] * V[f * H + h];
    out[i] = s;
}
// a_src[n,h], a_dst[n,h] from hx
__global__ void k_a_srcdst(const float* __restrict__ hx, const float* __restrict__ as,
                           const float* __restrict__ ad, float* __restrict__ osrc,
                           float* __restrict__ odst, int H) {
    int i = blockIdx.x * blockDim.x + threadIdx.x;
    if (i >= Nn * H) return;
    int n = i / H, h = i % H;
    const float* row = hx + (long)n * (H * Cc) + h * Cc;
    float s = 0.f, d = 0.f;
    for (int c = 0; c < Cc; ++c) { s += row[c] * as[h * Cc + c]; d += row[c] * ad[h * Cc + c]; }
    osrc[i] = s; odst[i] = d;
}

// ------------------------- segment softmax passes ---------------------------
__global__ void k_edge_max(const int* __restrict__ src, const int* __restrict__ dst,
                           const float* __restrict__ asv, const float* __restrict__ adv,
                           float* __restrict__ aed, float* __restrict__ m, int H) {
    int i = blockIdx.x * blockDim.x + threadIdx.x;
    if (i >= Ee * H) return;
    int e = i / H, h = i % H;
    float a = lrelu(asv[src[e] * H + h] + adv[dst[e] * H + h] + aed[i], 0.2f);
    aed[i] = a;
    atomicMaxF(&m[dst[e] * H + h], a);
}
__global__ void k_self_max(const float* __restrict__ asv, const float* __restrict__ adv,
                           float* __restrict__ asl, float* __restrict__ m, int H) {
    int i = blockIdx.x * blockDim.x + threadIdx.x;
    if (i >= Nn * H) return;
    float a = lrelu(asv[i] + adv[i] + asl[i], 0.2f);
    asl[i] = a;
    atomicMaxF(&m[i], a);
}
__global__ void k_edge_exp(const int* __restrict__ dst, float* __restrict__ aed,
                           const float* __restrict__ m, float* __restrict__ denom, int H) {
    int i = blockIdx.x * blockDim.x + threadIdx.x;
    if (i >= Ee * H) return;
    int e = i / H, h = i % H;
    float ex = __expf(aed[i] - m[dst[e] * H + h]);
    aed[i] = ex;
    atomAddF(&denom[dst[e] * H + h], ex);
}
__global__ void k_self_exp(float* __restrict__ asl, const float* __restrict__ m,
                           float* __restrict__ denom, int H) {
    int i = blockIdx.x * blockDim.x + threadIdx.x;
    if (i >= Nn * H) return;
    float ex = __expf(asl[i] - m[i]);
    asl[i] = ex;
    atomAddF(&denom[i], ex);
}

// ------------------------------ aggregation ---------------------------------
// one wave per edge; lanes cover 32 channels of each head
__global__ void k_edge_agg(const int* __restrict__ src, const int* __restrict__ dst,
                           const float* __restrict__ hx, const float* __restrict__ aed,
                           const float* __restrict__ denom, float* __restrict__ out, int H) {
    int e = blockIdx.x * (blockDim.x >> 5) + (threadIdx.x >> 5);
    int lane = threadIdx.x & 31;
    if (e >= Ee) return;
    int s = src[e], d = dst[e];
    int HCl = H * Cc;
    for (int h = 0; h < H; ++h) {
        float w = aed[e * H + h] / denom[d * H + h];
        int j = h * Cc + lane;
        atomAddF(&out[(long)d * HCl + j], hx[(long)s * HCl + j] * w);
    }
}
__global__ void k_self_agg(const float* __restrict__ hx, const float* __restrict__ asl,
                           const float* __restrict__ denom, float* __restrict__ out, int H) {
    int HCl = H * Cc;
    int i = blockIdx.x * blockDim.x + threadIdx.x;
    if (i >= Nn * HCl) return;
    int n = i / HCl, h = (i % HCl) / Cc;
    float w = asl[n * H + h] / denom[n * H + h];
    atomAddF(&out[i], hx[i] * w);
}

// bias + (optional) leaky_relu 0.01; optionally emit bf16 activations for next GEMM
__global__ void k_bias_act(float* __restrict__ out, const float* __restrict__ bias,
                           __bf16* __restrict__ xb, int HCl, int act) {
    int i = blockIdx.x * blockDim.x + threadIdx.x;
    if (i >= Nn * HCl) return;
    float v = out[i] + bias[i % HCl];
    if (act) v = lrelu(v, 0.01f);
    out[i] = v;
    if (xb) xb[i] = (__bf16)v;
}

// ------------------------------- pooling ------------------------------------
__global__ void k_cnt(const int* __restrict__ batch, int* __restrict__ cnt) {
    int n = blockIdx.x * blockDim.x + threadIdx.x;
    if (n < Nn) atomicAdd(&cnt[batch[n]], 1);
}
__global__ void k_pool(const int* __restrict__ batch, const float* __restrict__ x,
                       float* __restrict__ hg) {
    int i = blockIdx.x * blockDim.x + threadIdx.x;
    if (i >= Nn * Cc) return;
    int n = i / Cc, c = i % Cc;
    atomAddF(&hg[batch[n] * Cc + c], x[i]);
}
__global__ void k_pool_div(float* __restrict__ hg, const int* __restrict__ cnt) {
    int i = blockIdx.x * blockDim.x + threadIdx.x;
    if (i >= Gg * Cc) return;
    hg[i] /= fmaxf((float)cnt[i / Cc], 1.f);
}

// --------------------------------- MLP --------------------------------------
__global__ void k_linear(const float* __restrict__ in, const float* __restrict__ W,
                         const float* __restrict__ b, float* __restrict__ out,
                         int fin, int fout, int act) {
    int i = blockIdx.x * blockDim.x + threadIdx.x;
    if (i >= Gg * fout) return;
    int g = i / fout, j = i % fout;
    float s = b[j];
    for (int k = 0; k < fin; ++k) s += in[g * fin + k] * W[k * fout + j];
    out[i] = act ? lrelu(s, 0.01f) : s;
}
__global__ void k_head(const float* __restrict__ in, const float* __restrict__ W0,
                       const float* __restrict__ b0, const float* __restrict__ W1,
                       const float* __restrict__ b1, const int* __restrict__ task,
                       float* __restrict__ out, int ncols) {
    int i = blockIdx.x * blockDim.x + threadIdx.x;
    if (i >= Gg * ncols) return;
    int g = i / ncols, j = i % ncols;
    const float* W = (*task == 0) ? W0 : W1;
    const float* b = (*task == 0) ? b0 : b1;
    float s = b[j];
    for (int k = 0; k < 128; ++k) s += in[g * 128 + k] * W[k * ncols + j];
    out[i] = s;
}

// ------------------------------- launcher -----------------------------------
static inline unsigned cdiv(long long a, long long b) { return (unsigned)((a + b - 1) / b); }

extern "C" void kernel_launch(void* const* d_in, const int* in_sizes, int n_in,
                              void* d_out, int out_size, void* d_ws, size_t ws_size,
                              hipStream_t stream) {
    (void)in_sizes; (void)n_in; (void)ws_size;
    const float* x     = (const float*)d_in[0];
    const int*   src   = (const int*)d_in[1];
    const int*   dst   = src + Ee;
    const float* eattr = (const float*)d_in[2];
    const int*   batch = (const int*)d_in[3];
    const int*   task  = (const int*)d_in[4];

    const float* Wl[3]  = {(const float*)d_in[5],  (const float*)d_in[11], (const float*)d_in[17]};
    const float* Wel[3] = {(const float*)d_in[6],  (const float*)d_in[12], (const float*)d_in[18]};
    const float* asl_[3]= {(const float*)d_in[7],  (const float*)d_in[13], (const float*)d_in[19]};
    const float* adl[3] = {(const float*)d_in[8],  (const float*)d_in[14], (const float*)d_in[20]};
    const float* ael[3] = {(const float*)d_in[9],  (const float*)d_in[15], (const float*)d_in[21]};
    const float* bl[3]  = {(const float*)d_in[10], (const float*)d_in[16], (const float*)d_in[22]};
    const float* lW1 = (const float*)d_in[23]; const float* lb1 = (const float*)d_in[24];
    const float* lW2 = (const float*)d_in[25]; const float* lb2 = (const float*)d_in[26];
    const float* oW0 = (const float*)d_in[27]; const float* ob0 = (const float*)d_in[28];
    const float* oW1 = (const float*)d_in[29]; const float* ob1 = (const float*)d_in[30];

    // ---- workspace carve-out ----
    char* base = (char*)d_ws; size_t off = 0;
    auto carve = [&](size_t bytes) -> void* {
        off = (off + 255) & ~(size_t)255;
        void* p = base + off; off += bytes; return p;
    };
    __bf16* xb    = (__bf16*)carve((size_t)Nn * FIN * 2);
    __bf16* wb    = (__bf16*)carve((size_t)FIN * FIN * 2);   // packed B
    float*  hx    = (float*)carve((size_t)Nn * FIN * 4);
    float*  outb  = (float*)carve((size_t)Nn * FIN * 4);
    float*  asv   = (float*)carve((size_t)Nn * 4 * 4);
    float*  adv   = (float*)carve((size_t)Nn * 4 * 4);
    float*  aed   = (float*)carve((size_t)Ee * 4 * 4);
    float*  aslf  = (float*)carve((size_t)Nn * 4 * 4);
    float*  mbuf  = (float*)carve((size_t)Nn * 4 * 4);
    float*  denom = (float*)carve((size_t)Nn * 4 * 4);
    float*  lattr = (float*)carve((size_t)Nn * FE * 4);
    int*    deg   = (int*)carve((size_t)Nn * 4);
    float*  Vbuf  = (float*)carve((size_t)FE * 4 * 4);
    float*  hg    = (float*)carve((size_t)Gg * Cc * 4);
    int*    cnt   = (int*)carve((size_t)Gg * 4);
    float*  t1    = (float*)carve((size_t)Gg * 128 * 4);
    float*  t2    = (float*)carve((size_t)Gg * 128 * 4);

    const int TB = 256;

    // ---- self-loop edge attrs (mean of incoming), computed once ----
    k_fill_i32<<<cdiv(Nn, TB), TB, 0, stream>>>(deg, 0, Nn);
    k_fill_f32<<<cdiv((long long)Nn * FE, TB), TB, 0, stream>>>(lattr, 0.f, Nn * FE);
    k_deg<<<cdiv(Ee, TB), TB, 0, stream>>>(dst, deg);
    k_loop_acc<<<cdiv((long long)Ee * FE, TB), TB, 0, stream>>>(dst, eattr, lattr);
    k_loop_div<<<cdiv((long long)Nn * FE, TB), TB, 0, stream>>>(lattr, deg);

    // ---- input activations to bf16 ----
    k_cvt_bf16<<<cdiv((long long)Nn * FIN, TB), TB, 0, stream>>>(x, xb, Nn * FIN);

    const int Hs[3] = {4, 4, 1};
    for (int l = 0; l < 3; ++l) {
        int H = Hs[l], HCl = H * Cc;

        // pack weights to WMMA lane layout, then hx = x @ W on WMMA
        k_pack_B<<<cdiv((long long)FIN * HCl, TB), TB, 0, stream>>>(Wl[l], wb, HCl);
        unsigned gblocks = cdiv(Nn / 16, TB / 32);
        if (HCl == 128) k_gemm_bf16<8><<<gblocks, TB, 0, stream>>>(xb, wb, hx, Nn);
        else            k_gemm_bf16<2><<<gblocks, TB, 0, stream>>>(xb, wb, hx, Nn);

        // attention logit pieces
        k_edge_vec<<<1, 64, 0, stream>>>(Wel[l], ael[l], Vbuf, H);
        k_a_edge<<<cdiv((long long)Ee * H, TB), TB, 0, stream>>>(eattr, Vbuf, aed, Ee, H);
        k_a_edge<<<cdiv((long long)Nn * H, TB), TB, 0, stream>>>(lattr, Vbuf, aslf, Nn, H);
        k_a_srcdst<<<cdiv((long long)Nn * H, TB), TB, 0, stream>>>(hx, asl_[l], adl[l], asv, adv, H);

        // segment softmax + aggregate
        k_fill_f32<<<cdiv((long long)Nn * H, TB), TB, 0, stream>>>(mbuf, -INFINITY, Nn * H);
        k_fill_f32<<<cdiv((long long)Nn * H, TB), TB, 0, stream>>>(denom, 0.f, Nn * H);
        k_fill_f32<<<cdiv((long long)Nn * HCl, TB), TB, 0, stream>>>(outb, 0.f, Nn * HCl);
        k_edge_max<<<cdiv((long long)Ee * H, TB), TB, 0, stream>>>(src, dst, asv, adv, aed, mbuf, H);
        k_self_max<<<cdiv((long long)Nn * H, TB), TB, 0, stream>>>(asv, adv, aslf, mbuf, H);
        k_edge_exp<<<cdiv((long long)Ee * H, TB), TB, 0, stream>>>(dst, aed, mbuf, denom, H);
        k_self_exp<<<cdiv((long long)Nn * H, TB), TB, 0, stream>>>(aslf, mbuf, denom, H);
        k_edge_agg<<<cdiv(Ee, TB / 32), TB, 0, stream>>>(src, dst, hx, aed, denom, outb, H);
        k_self_agg<<<cdiv((long long)Nn * HCl, TB), TB, 0, stream>>>(hx, aslf, denom, outb, H);

        // bias + inter-layer activation; emit next layer's bf16 input
        k_bias_act<<<cdiv((long long)Nn * HCl, TB), TB, 0, stream>>>(
            outb, bl[l], (l < 2) ? xb : (__bf16*)nullptr, HCl, (l < 2) ? 1 : 0);
    }

    // ---- global mean pool over graphs ----
    k_fill_i32<<<cdiv(Gg, TB), TB, 0, stream>>>(cnt, 0, Gg);
    k_fill_f32<<<cdiv((long long)Gg * Cc, TB), TB, 0, stream>>>(hg, 0.f, Gg * Cc);
    k_cnt<<<cdiv(Nn, TB), TB, 0, stream>>>(batch, cnt);
    k_pool<<<cdiv((long long)Nn * Cc, TB), TB, 0, stream>>>(batch, outb, hg);
    k_pool_div<<<cdiv((long long)Gg * Cc, TB), TB, 0, stream>>>(hg, cnt);

    // ---- MLP head (tiny) ----
    k_linear<<<cdiv((long long)Gg * 128, TB), TB, 0, stream>>>(hg, lW1, lb1, t1, Cc, 128, 1);
    k_linear<<<cdiv((long long)Gg * 128, TB), TB, 0, stream>>>(t1, lW2, lb2, t2, 128, 128, 1);
    int ncols = out_size / Gg;
    k_head<<<cdiv((long long)Gg * ncols, TB), TB, 0, stream>>>(t2, oW0, ob0, oW1, ob1, task,
                                                               (float*)d_out, ncols);
}
